// GraphMatrixCompletion_37254546326131
// MI455X (gfx1250) — matrix operand order repaired and checked by reading, not compiled, not run on hardware
//
#include <hip/hip_runtime.h>

// ---- problem constants (match reference setup_inputs) ----
#define NU     100000
#define NV     50000
#define KSUP   5
#define ESUP   400000
#define NE     500000
#define DIN    512
#define FK     100      // D_GCN / K
#define DGCN   500
#define DSIDE  128
#define HSIDE  64
#define DCAT   564      // D_GCN + H_SIDE
#define DCATP  576      // padded to multiple of 32
#define DENC   128
#define NBASIS 3
#define NCLS   5

typedef __attribute__((ext_vector_type(16))) __bf16 v16bf;
typedef __attribute__((ext_vector_type(8)))  float  v8f;

union FragBF {
  v16bf v;
  unsigned u[8];
  uint4 q[2];
};

// pack two fp32 -> packed bf16 (truncate) with one v_perm_b32
__device__ __forceinline__ unsigned pack_bf16(float lo, float hi) {
  return __builtin_amdgcn_perm(__float_as_uint(hi), __float_as_uint(lo), 0x07060302u);
}
__device__ __forceinline__ unsigned short trunc_bf16(float f) {
  return (unsigned short)(__float_as_uint(f) >> 16);
}

// Build the 16-bit A-matrix 16x32 fragment for this lane from a row pointer.
// Lanes 0-15 hold K {kb..kb+7, kb+16..kb+23} with kb=0; lanes 16-31 kb=8.
__device__ __forceinline__ void load_a_frag(FragBF& a, const float* __restrict__ arow,
                                            int k0, int kb) {
  float4 f0 = *(const float4*)(arow + k0 + kb);
  float4 f1 = *(const float4*)(arow + k0 + kb + 4);
  float4 f2 = *(const float4*)(arow + k0 + kb + 16);
  float4 f3 = *(const float4*)(arow + k0 + kb + 20);
  a.u[0] = pack_bf16(f0.x, f0.y);  a.u[1] = pack_bf16(f0.z, f0.w);
  a.u[2] = pack_bf16(f1.x, f1.y);  a.u[3] = pack_bf16(f1.z, f1.w);
  a.u[4] = pack_bf16(f2.x, f2.y);  a.u[5] = pack_bf16(f2.z, f2.w);
  a.u[6] = pack_bf16(f3.x, f3.y);  a.u[7] = pack_bf16(f3.z, f3.w);
}

// ============================================================================
// WMMA GEMM: C[z] = act( A[z] (rows optionally gathered) * B[z] + bias )
// fp32 in/out, bf16 compute. Block = 256 threads = 8 waves.
// Wave w owns a 16(M) x 64(N) C tile: one A fragment feeds 4 WMMAs per K-step.
// B tile (32x64) staged in LDS as bf16, layout Bs[n][k].
// Requires: Kp % 32 == 0, M % 16 == 0 (true at all call sites); N guarded.
// ============================================================================
__global__ __launch_bounds__(256)
void gemm_wmma(const float* __restrict__ A, int lda, long long strideA,
               const float* __restrict__ B, int ldb, long long strideB,
               float* __restrict__ C, int ldc, long long strideC,
               int N, int Kp, int Kreal,
               const float* __restrict__ bias, int do_relu,
               const int* __restrict__ idxA, int mtiles) {
  __shared__ __align__(16) unsigned short Bs[64 * 32];  // [n][k], 4 KB

  const int z = blockIdx.z;
  A += strideA * z;  B += strideB * z;  C += strideC * z;

  const int lane  = threadIdx.x & 31;
  const int wave  = threadIdx.x >> 5;
  const int n0    = blockIdx.x * 64;
  const int mtile = blockIdx.y * 8 + wave;
  const bool active = (mtile < mtiles);

  const float* arow = nullptr;
  if (active) {
    int m = mtile * 16 + (lane & 15);
    int r = idxA ? idxA[m] : m;
    arow = A + (long long)r * lda;
  }
  const int kb = (lane < 16) ? 0 : 8;
  const int bofs = (lane & 15) * 32 + ((lane >> 4) << 3);

  v8f acc[4] = {};
  for (int k0 = 0; k0 < Kp; k0 += 32) {
    __syncthreads();
    // stage B tile (32 x 64) -> LDS bf16, layout [n][k]
    for (int e = threadIdx.x; e < 2048; e += 256) {
      int n = e >> 5, k = e & 31;
      float v = 0.f;
      if ((k0 + k) < Kreal && (n0 + n) < N)
        v = B[(long long)(k0 + k) * ldb + (n0 + n)];
      Bs[n * 32 + k] = trunc_bf16(v);
    }
    __syncthreads();

    if (active) {
      FragBF a;
      load_a_frag(a, arow, k0, kb);
#pragma unroll
      for (int nt = 0; nt < 4; ++nt) {
        FragBF b;
        const uint4* bp = (const uint4*)&Bs[nt * 16 * 32 + bofs];
        b.q[0] = bp[0];
        b.q[1] = bp[2];
        acc[nt] = __builtin_amdgcn_wmma_f32_16x16x32_bf16(
            false, a.v, false, b.v, (short)0, acc[nt], false, false);
      }
    }
  }

  if (active) {
    int rb = mtile * 16 + ((lane < 16) ? 0 : 8);
#pragma unroll
    for (int nt = 0; nt < 4; ++nt) {
      int n = n0 + nt * 16 + (lane & 15);
      if (n < N) {
        float bv = bias ? bias[n] : 0.f;
#pragma unroll
        for (int r = 0; r < 8; ++r) {
          float v = acc[nt][r] + bv;
          if (do_relu) v = v > 0.f ? v : 0.f;
          C[(long long)(rb + r) * ldc + n] = v;
        }
      }
    }
  }
}

// ============================================================================
// Fused decoder: out[e] = (sum_k (u_e . W_b . v_e) basis) @ dec_cls, no scratch.
// Wave owns 16 edges. A (gathered user embeds, 16x128) preloaded as 4 bf16
// K-fragments. dec_W[b] (128x128) staged in LDS as bf16 once per block.
// 8 N-tiles x 4 K-steps of WMMA per basis; C tile consumed in-register against
// gathered item rows; per-row dot reduced across each 16-lane half via shfl.
// ============================================================================
__global__ __launch_bounds__(256)
void decoder_fused(const float* __restrict__ uemb, const float* __restrict__ iemb,
                   const int* __restrict__ ue_idx, const int* __restrict__ ie_idx,
                   const float* __restrict__ dec_W, const float* __restrict__ dec_cls,
                   float* __restrict__ out, int mtiles) {
  __shared__ __align__(16) unsigned short Bs[DENC * DENC];  // 32 KB, [n][k]

  const int lane  = threadIdx.x & 31;
  const int wave  = threadIdx.x >> 5;
  const int mtile = blockIdx.x * 8 + wave;
  const bool active = (mtile < mtiles);
  const int kb   = (lane < 16) ? 0 : 8;
  const int half = (lane < 16) ? 0 : 8;

  // preload gathered A fragments (all 4 K-steps), reused for all 3 bases
  FragBF af[4];
  const float* vr[8];
  if (active) {
    int m = mtile * 16 + (lane & 15);
    const float* ar = uemb + (long long)ue_idx[m] * DENC;
#pragma unroll
    for (int kk = 0; kk < 4; ++kk) load_a_frag(af[kk], ar, kk * 32, kb);
    int base = mtile * 16 + half;
#pragma unroll
    for (int r = 0; r < 8; ++r)
      vr[r] = iemb + (long long)ie_idx[base + r] * DENC;
  }

  float outacc[8 * NCLS] = {};

  for (int b = 0; b < NBASIS; ++b) {
    const float* Wb = dec_W + (long long)b * DENC * DENC;
    __syncthreads();
    // stage W_b -> LDS bf16, [n][k]; coalesced global reads
    for (int e = threadIdx.x; e < DENC * DENC; e += 256) {
      int n = e & 127, k = e >> 7;
      Bs[n * DENC + k] = trunc_bf16(Wb[k * DENC + n]);
    }
    __syncthreads();

    if (active) {
      float rowsum[8] = {};
#pragma unroll
      for (int nt = 0; nt < 8; ++nt) {
        v8f acc = {};
#pragma unroll
        for (int kk = 0; kk < 4; ++kk) {
          FragBF bfr;
          const uint4* bp = (const uint4*)
              &Bs[(nt * 16 + (lane & 15)) * DENC + kk * 32 + ((lane >> 4) << 3)];
          bfr.q[0] = bp[0];
          bfr.q[1] = bp[2];
          acc = __builtin_amdgcn_wmma_f32_16x16x32_bf16(
              false, af[kk].v, false, bfr.v, (short)0, acc, false, false);
        }
        // consume: lane holds col = nt*16 + (lane&15), rows half+0..7
        int col = nt * 16 + (lane & 15);
#pragma unroll
        for (int r = 0; r < 8; ++r)
          rowsum[r] += acc[r] * vr[r][col];
      }
      // reduce over the 16 lanes (columns) of this half
      float dc[NCLS];
#pragma unroll
      for (int c = 0; c < NCLS; ++c) dc[c] = dec_cls[b * NCLS + c];
#pragma unroll
      for (int r = 0; r < 8; ++r) {
        float s = rowsum[r];
        s += __shfl_xor(s, 1, 32);
        s += __shfl_xor(s, 2, 32);
        s += __shfl_xor(s, 4, 32);
        s += __shfl_xor(s, 8, 32);
#pragma unroll
        for (int c = 0; c < NCLS; ++c) outacc[r * NCLS + c] += s * dc[c];
      }
    }
  }

  if (active && (lane & 15) == 0) {
    int base = mtile * 16 + half;
#pragma unroll
    for (int r = 0; r < 8; ++r)
#pragma unroll
      for (int c = 0; c < NCLS; ++c)
        out[(long long)(base + r) * NCLS + c] = outacc[r * NCLS + c];
  }
}

// ---- zero fill ----
__global__ void zero_f32(float* p, long long n) {
  long long stride = (long long)gridDim.x * blockDim.x;
  for (long long i = (long long)blockIdx.x * blockDim.x + threadIdx.x; i < n; i += stride)
    p[i] = 0.f;
}

// ---- sparse aggregation: one wave per edge, atomics into padded concat buf ----
__global__ __launch_bounds__(256)
void agg_user(const int* __restrict__ rows, const int* __restrict__ cols,
              const float* __restrict__ vals, const float* __restrict__ tmpv,
              float* __restrict__ ucat) {
  int gw   = blockIdx.x * 8 + (threadIdx.x >> 5);   // edge over K*ESUP
  int lane = threadIdx.x & 31;
  if (gw >= KSUP * ESUP) return;
  int k = gw / ESUP, e = gw - k * ESUP;
  long long o = (long long)k * ESUP + e;
  int u = rows[o], c = cols[o];
  float val = vals[o];
  const float* src = tmpv + ((long long)k * NV + c) * FK;
  float* dst = ucat + (long long)u * DCATP + k * FK;
  for (int f = lane; f < FK; f += 32)
    unsafeAtomicAdd(&dst[f], val * src[f]);
}

__global__ __launch_bounds__(256)
void agg_item(const int* __restrict__ rows, const int* __restrict__ cols,
              const float* __restrict__ vals, const float* __restrict__ tmpu,
              float* __restrict__ icat) {
  int gw   = blockIdx.x * 8 + (threadIdx.x >> 5);
  int lane = threadIdx.x & 31;
  if (gw >= KSUP * ESUP) return;
  int k = gw / ESUP, e = gw - k * ESUP;
  long long o = (long long)k * ESUP + e;
  int u = rows[o], c = cols[o];
  float val = vals[o];
  const float* src = tmpu + ((long long)k * NU + u) * FK;
  float* dst = icat + (long long)c * DCATP + k * FK;
  for (int f = lane; f < FK; f += 32)
    unsafeAtomicAdd(&dst[f], val * src[f]);
}

// ---- in-place relu on first `cols` columns of a [rows, ld] matrix ----
__global__ void relu_cols(float* p, long long rows, int cols, int ld) {
  long long total = rows * cols;
  long long stride = (long long)gridDim.x * blockDim.x;
  for (long long i = (long long)blockIdx.x * blockDim.x + threadIdx.x; i < total; i += stride) {
    long long r = i / cols;
    int c = (int)(i - r * cols);
    float* q = p + r * ld + c;
    float v = *q;
    *q = v > 0.f ? v : 0.f;
  }
}

extern "C" void kernel_launch(void* const* d_in, const int* in_sizes, int n_in,
                              void* d_out, int out_size, void* d_ws, size_t ws_size,
                              hipStream_t stream) {
  (void)in_sizes; (void)n_in; (void)out_size; (void)ws_size;
  const int*   sup_u   = (const int*)  d_in[0];
  const int*   sup_i   = (const int*)  d_in[1];
  const float* sup_v   = (const float*)d_in[2];
  const float* uin     = (const float*)d_in[3];
  const float* vin     = (const float*)d_in[4];
  const float* uside   = (const float*)d_in[5];
  const float* vside   = (const float*)d_in[6];
  const int*   ue_idx  = (const int*)  d_in[7];
  const int*   ie_idx  = (const int*)  d_in[8];
  const float* W_enc   = (const float*)d_in[9];
  const float* Wu1     = (const float*)d_in[10];
  const float* bu1     = (const float*)d_in[11];
  const float* Wi1     = (const float*)d_in[12];
  const float* bi1     = (const float*)d_in[13];
  const float* Wu2     = (const float*)d_in[14];
  const float* Wi2     = (const float*)d_in[15];
  const float* dec_W   = (const float*)d_in[16];
  const float* dec_cls = (const float*)d_in[17];

  // ---- workspace layout (bytes) ----
  char* ws = (char*)d_ws;
  float* tmp_u = (float*)(ws);                        // [5, NU, 100]  200 MB
  float* tmp_v = (float*)(ws + 200000000LL);          // [5, NV, 100]  100 MB
  float* ucat  = (float*)(ws + 300000000LL);          // [NU, 576]     230.4 MB
  float* icat  = (float*)(ws + 530400000LL);          // [NV, 576]     115.2 MB
  float* uemb  = (float*)(ws + 645600000LL);          // [NU, 128]     51.2 MB
  float* iemb  = (float*)(ws + 696800000LL);          // [NV, 128]     25.6 MB

  dim3 blk(256);

  // zero accumulators + pad columns
  zero_f32<<<4096, blk, 0, stream>>>(ucat, (long long)NU * DCATP);
  zero_f32<<<4096, blk, 0, stream>>>(icat, (long long)NV * DCATP);

  // ---- encoder GEMMs: tmp[k] = inputs @ W_enc[k]  (batched over gridDim.z) ----
  gemm_wmma<<<dim3(2, (NU/16 + 7)/8, KSUP), blk, 0, stream>>>(
      uin, DIN, 0LL, W_enc, FK, (long long)DIN * FK,
      tmp_u, FK, (long long)NU * FK,
      FK, DIN, DIN, nullptr, 0, nullptr, NU / 16);
  gemm_wmma<<<dim3(2, (NV/16 + 7)/8, KSUP), blk, 0, stream>>>(
      vin, DIN, 0LL, W_enc, FK, (long long)DIN * FK,
      tmp_v, FK, (long long)NV * FK,
      FK, DIN, DIN, nullptr, 0, nullptr, NV / 16);

  // ---- sparse aggregation (fused transpose/concat: col = k*100 + f) ----
  agg_user<<<(KSUP * ESUP) / 8, blk, 0, stream>>>(sup_u, sup_i, sup_v, tmp_v, ucat);
  agg_item<<<(KSUP * ESUP) / 8, blk, 0, stream>>>(sup_u, sup_i, sup_v, tmp_u, icat);

  // relu on GCN part only
  relu_cols<<<4096, blk, 0, stream>>>(ucat, NU, DGCN, DCATP);
  relu_cols<<<4096, blk, 0, stream>>>(icat, NV, DGCN, DCATP);

  // ---- side dense (bias + relu) written into concat cols [500, 564) ----
  gemm_wmma<<<dim3(1, (NU/16 + 7)/8, 1), blk, 0, stream>>>(
      uside, DSIDE, 0LL, Wu1, HSIDE, 0LL,
      ucat + DGCN, DCATP, 0LL,
      HSIDE, DSIDE, DSIDE, bu1, 1, nullptr, NU / 16);
  gemm_wmma<<<dim3(1, (NV/16 + 7)/8, 1), blk, 0, stream>>>(
      vside, DSIDE, 0LL, Wi1, HSIDE, 0LL,
      icat + DGCN, DCATP, 0LL,
      HSIDE, DSIDE, DSIDE, bi1, 1, nullptr, NV / 16);

  // ---- dense2: embed = cat @ W2   (Kp=576 padded, Kreal=564) ----
  gemm_wmma<<<dim3(2, (NU/16 + 7)/8, 1), blk, 0, stream>>>(
      ucat, DCATP, 0LL, Wu2, DENC, 0LL,
      uemb, DENC, 0LL,
      DENC, DCATP, DCAT, nullptr, 0, nullptr, NU / 16);
  gemm_wmma<<<dim3(2, (NV/16 + 7)/8, 1), blk, 0, stream>>>(
      icat, DCATP, 0LL, Wi2, DENC, 0LL,
      iemb, DENC, 0LL,
      DENC, DCATP, DCAT, nullptr, 0, nullptr, NV / 16);

  // ---- fully fused decoder: gathers, 3 bilinear bases, dec_cls epilogue ----
  decoder_fused<<<dim3((NE/16 + 7)/8), blk, 0, stream>>>(
      uemb, iemb, ue_idx, ie_idx, dec_W, dec_cls, (float*)d_out, NE / 16);
}
